// VAECW_52295521796295
// MI455X (gfx1250) — compile-verified, matrix operand order checked by reading, not run on hardware
//
#include <hip/hip_runtime.h>
#include <hip/hip_bf16.h>

typedef __attribute__((ext_vector_type(16))) _Float16 v16h;
typedef __attribute__((ext_vector_type(8)))  _Float16 v8h;
typedef __attribute__((ext_vector_type(8)))  float    v8f;

#define BB 512
#define DD 256
#define EE 16
#define KK 512
#define ZZ 64
#define HH 512
#define PP 400
#define WDIM 4096
#define MROWS (BB + PP)   // 912

// ---- d_out flat offsets (floats), in reference return order ----
#define OFF_MU    0
#define OFF_LV    (OFF_MU   + BB*ZZ)          // 32768
#define OFF_PMU   (OFF_LV   + BB*ZZ)          // 65536
#define OFF_PLV   (OFF_PMU  + PP*ZZ)          // 91136
#define OFF_Z     (OFF_PLV  + PP*ZZ)          // 116736
#define OFF_RECON (OFF_Z    + BB*ZZ)          // 149504
#define OFF_D     (OFF_RECON+ BB*WDIM)        // 2246656
#define OFF_IDX   (OFF_D    + (long)BB*DD*KK) // 69355520

union FragU { v16h v; v8h h[2]; };

// ---------------------------------------------------------------------------
// prep kernels
// ---------------------------------------------------------------------------

// build xa16 (912 x 4096) f16:
//   rows [0,512): xa[b][e*256+d] = x[b][d*16+e]   (the (B,D,E)->(B,E,D) transpose)
//   rows [512,912): straight convert of pseudo_inputs (already (E,D) layout)
__global__ void pack_xa_f16(const float* __restrict__ x,
                            const float* __restrict__ pseudo,
                            _Float16* __restrict__ xa)
{
    long t = (long)blockIdx.x * blockDim.x + threadIdx.x;
    if (t >= (long)MROWS * WDIM) return;
    int row = (int)(t / WDIM);
    int i   = (int)(t % WDIM);
    float v;
    if (row < BB) {
        int e = i / DD, d = i % DD;
        v = x[(long)row * WDIM + d * EE + e];
    } else {
        v = pseudo[(long)(row - BB) * WDIM + i];
    }
    xa[t] = (_Float16)v;
}

// dst[c*R + r] = (f16) src[r*C + c]   (weights: row-major (K x N) -> col-major f16)
__global__ void transpose_f32_to_f16(const float* __restrict__ src,
                                     _Float16* __restrict__ dst, int R, int C)
{
    long t = (long)blockIdx.x * blockDim.x + threadIdx.x;
    if (t >= (long)R * C) return;
    int r = (int)(t / C);
    int c = (int)(t % C);
    dst[(long)c * R + r] = (_Float16)src[t];
}

// codebook (D,K,E) f32 -> f16 copy (same layout == col-major B for dist GEMM)
// plus cc[d*K+k] = sum_e cb^2
__global__ void prep_codebook(const float* __restrict__ cb,
                              _Float16* __restrict__ cb16,
                              float* __restrict__ cc)
{
    long t = (long)blockIdx.x * blockDim.x + threadIdx.x;   // over D*K
    if (t >= (long)DD * KK) return;
    const float* p = cb + t * EE;
    float s = 0.f;
    #pragma unroll
    for (int e = 0; e < EE; ++e) {
        float v = p[e];
        cb16[t * EE + e] = (_Float16)v;
        s += v * v;
    }
    cc[t] = s;
}

// rr[b*D+d] = sum_e recon^2 (reads f32 recon already written to d_out)
__global__ void prep_row_norms(const float* __restrict__ recon,
                               float* __restrict__ rr)
{
    long t = (long)blockIdx.x * blockDim.x + threadIdx.x;   // over B*D
    if (t >= (long)BB * DD) return;
    const float* p = recon + t * EE;
    float s = 0.f;
    #pragma unroll
    for (int e = 0; e < EE; ++e) { float v = p[e]; s += v * v; }
    rr[t] = s;
}

// packed (key<<32 | k) argmin accumulator init to +inf
__global__ void init_packed(unsigned long long* __restrict__ p)
{
    long t = (long)blockIdx.x * blockDim.x + threadIdx.x;
    if (t < (long)BB * DD) p[t] = ~0ull;
}

__global__ void finalize_idx(const unsigned long long* __restrict__ p,
                             float* __restrict__ idx_out)
{
    long t = (long)blockIdx.x * blockDim.x + threadIdx.x;
    if (t < (long)BB * DD) idx_out[t] = (float)(unsigned)(p[t] & 0xFFFFFFFFull);
}

// ---------------------------------------------------------------------------
// WMMA GEMM: C[M,N] = act(A16[MxK] * B16t[NxK] + bias)
//   A row-major f16 (lda = K), Bt row-major f16 (ldb = K) == B column-major.
//   one wave -> one 16x64 tile (4 accumulators, A fragment reused 4x).
//   All fragments for a k-step are loaded before the WMMA chain so the
//   compiler can retire loads with partial s_wait_loadcnt values.
//   blockDim = 128 (4 waves). M % 16 == 0, N % 64 == 0, K % 32 == 0.
// ---------------------------------------------------------------------------
__global__ void gemm_wmma_f16(const _Float16* __restrict__ A, int lda,
                              const _Float16* __restrict__ Bt, int ldb,
                              const float* __restrict__ bias,
                              float* __restrict__ Cf32,
                              _Float16* __restrict__ Cf16,
                              int ldc, int M, int N, int Kd, int relu)
{
    const int wave = threadIdx.x >> 5;
    const int lane = threadIdx.x & 31;
    const int n0 = (blockIdx.x * (blockDim.x >> 5) + wave) * 64;
    const int m0 = blockIdx.y * 16;
    if (n0 >= N || m0 >= M) return;           // uniform per wave

    const int mm = lane & 15;                 // M row (A) / N col (B)
    const int hi = lane >> 4;                 // K sub-chunk select
    const _Float16* pa = A  + (long)(m0 + mm) * lda + hi * 8;
    const _Float16* pb = Bt + (long)(n0 + mm) * ldb + hi * 8;
    const long nstep = (long)16 * ldb;        // 16 columns of B

    v8f acc[4];
    #pragma unroll
    for (int j = 0; j < 4; ++j) acc[j] = (v8f){};

    for (int k = 0; k < Kd; k += 32) {
        FragU ua, ub[4];
        // issue all loads for this k-step up front
        ua.h[0] = *(const v8h*)(pa + k);      // K = k + hi*8 + [0..7]
        ua.h[1] = *(const v8h*)(pa + k + 16); // K = k + 16 + hi*8 + [0..7]
        #pragma unroll
        for (int j = 0; j < 4; ++j) {
            ub[j].h[0] = *(const v8h*)(pb + j * nstep + k);
            ub[j].h[1] = *(const v8h*)(pb + j * nstep + k + 16);
        }
        // WGP-scope prefetch two k-steps ahead (locality 3 -> near caches)
        __builtin_prefetch(pa + k + 64, 0, 3);
        #pragma unroll
        for (int j = 0; j < 4; ++j)
            __builtin_prefetch(pb + j * nstep + k + 64, 0, 3);
        #pragma unroll
        for (int j = 0; j < 4; ++j)
            acc[j] = __builtin_amdgcn_wmma_f32_16x16x32_f16(
                         false, ua.v, false, ub[j].v, (short)0, acc[j], false, false);
    }

    const int n = lane & 15;
    #pragma unroll
    for (int j = 0; j < 4; ++j) {
        int nc = n0 + j * 16 + n;
        float bv = bias ? bias[nc] : 0.f;
        #pragma unroll
        for (int r = 0; r < 8; ++r) {
            int m = r + 8 * hi;               // C layout: VGPR r -> M = r (+8 hi half)
            float v = acc[j][r] + bv;
            if (relu) v = fmaxf(v, 0.f);
            long o = (long)(m0 + m) * ldc + nc;
            if (Cf32) Cf32[o] = v;
            if (Cf16) Cf16[o] = (_Float16)v;
        }
    }
}

// ---------------------------------------------------------------------------
// reparameterization + output of mu/log_var/pseudo_*/z; z16 for decoder
// ---------------------------------------------------------------------------
__device__ __forceinline__ unsigned pcg_hash(unsigned v)
{
    unsigned s = v * 747796405u + 2891336453u;
    unsigned w = ((s >> ((s >> 28) + 4u)) ^ s) * 277803737u;
    return (w >> 22) ^ w;
}
__device__ __forceinline__ float rand_normal(unsigned i)
{
    unsigned u1 = pcg_hash(2u * i + 1u);
    unsigned u2 = pcg_hash(2u * i + 2u);
    float f1 = ((float)u1 + 1.0f) * (1.0f / 4294967296.0f);
    float f2 = (float)u2 * (1.0f / 4294967296.0f);
    return sqrtf(-2.0f * logf(f1)) * __cosf(6.28318530718f * f2);
}

__global__ void reparam(const float* __restrict__ enc_out, // (912 x 128)
                        float* __restrict__ out,
                        _Float16* __restrict__ z16)
{
    int t = blockIdx.x * blockDim.x + threadIdx.x;          // over 912*64
    if (t >= MROWS * ZZ) return;
    int b = t / ZZ, j = t % ZZ;
    float mu = enc_out[b * 2 * ZZ + j];
    float lv = enc_out[b * 2 * ZZ + ZZ + j];
    if (b < BB) {
        out[OFF_MU + b * ZZ + j] = mu;
        out[OFF_LV + b * ZZ + j] = lv;
        float z = mu + rand_normal((unsigned)(b * ZZ + j)) * expf(0.5f * lv);
        out[OFF_Z + b * ZZ + j] = z;
        z16[b * ZZ + j] = (_Float16)z;
    } else {
        int p = b - BB;
        out[OFF_PMU + p * ZZ + j] = mu;
        out[OFF_PLV + p * ZZ + j] = lv;
    }
}

// ---------------------------------------------------------------------------
// distance kernel + fused argmin:
//   d[b,d,k] = rr[b,d] + cc[d,k] - 2 * sum_e r[b,d,e]*cb[d,k,e]
// one wave per 16b x 64k tile per d (4 WMMAs, A fragment reused; K=16 padded
// to 32 with zeros).  tiles = D * (B/16) * (K/64) = 256*32*8 = 65536.
// per-row argmin: lane-local best over 4 k-subtiles, shfl_xor reduction
// across the 16-lane half, then GLOBAL_ATOMIC_MIN_U64 on (key<<32 | k).
// ---------------------------------------------------------------------------
__global__ void dist_wmma(const _Float16* __restrict__ recon16,
                          const _Float16* __restrict__ cb16,
                          const float* __restrict__ rr,
                          const float* __restrict__ cc,
                          float* __restrict__ dout,
                          unsigned long long* __restrict__ packed)
{
    const int wave = threadIdx.x >> 5;
    const int lane = threadIdx.x & 31;
    long tile = (long)blockIdx.x * (blockDim.x >> 5) + wave;
    const int kt = (int)(tile & 7);           // 64-wide k tile
    const int bt = (int)((tile >> 3) & 31);
    const int d  = (int)(tile >> 8);

    const int mm = lane & 15, hi = lane >> 4;
    const int n  = lane & 15;
    v8h zero = {};

    // issue all fragment loads up front
    FragU ua, ub[4];
    ua.h[0] = *(const v8h*)(recon16 + (long)(bt * 16 + mm) * WDIM + d * EE + hi * 8);
    ua.h[1] = zero;                           // K 16..31 padded
    #pragma unroll
    for (int j = 0; j < 4; ++j) {
        ub[j].h[0] = *(const v8h*)(cb16 + ((long)d * KK + kt * 64 + j * 16 + mm) * EE + hi * 8);
        ub[j].h[1] = zero;
    }

    float rv[8];
    #pragma unroll
    for (int r = 0; r < 8; ++r)
        rv[r] = rr[(long)(bt * 16 + r + 8 * hi) * DD + d];

    v8f c[4];
    #pragma unroll
    for (int j = 0; j < 4; ++j) {
        v8f cz = {};
        c[j] = __builtin_amdgcn_wmma_f32_16x16x32_f16(
                   false, ua.v, false, ub[j].v, (short)0, cz, false, false);
    }

    float bestv[8];
    int   bestk[8];
    #pragma unroll
    for (int r = 0; r < 8; ++r) { bestv[r] = 3.402823466e+38f; bestk[r] = 0; }

    #pragma unroll
    for (int j = 0; j < 4; ++j) {
        const int k = kt * 64 + j * 16 + n;
        const float ccv = cc[(long)d * KK + k];
        #pragma unroll
        for (int r = 0; r < 8; ++r) {
            int b = bt * 16 + r + 8 * hi;
            float v = rv[r] + ccv - 2.0f * c[j][r];
            dout[(long)b * DD * KK + (long)d * KK + k] = v;
            if (v < bestv[r] || (v == bestv[r] && k < bestk[r])) {
                bestv[r] = v; bestk[r] = k;
            }
        }
    }

    // per-row reduce across the 16-lane half (xor masks < 16 stay in half)
    #pragma unroll
    for (int r = 0; r < 8; ++r) {
        float bv = bestv[r];
        int   bk = bestk[r];
        #pragma unroll
        for (int m = 8; m >= 1; m >>= 1) {
            float ov = __shfl_xor(bv, m, 32);
            int   ok = __shfl_xor(bk, m, 32);
            if (ov < bv || (ov == bv && ok < bk)) { bv = ov; bk = ok; }
        }
        if ((lane & 15) == 0) {
            int b = bt * 16 + r + 8 * hi;
            unsigned bits = __float_as_uint(bv);
            unsigned key  = (bits & 0x80000000u) ? ~bits : (bits ^ 0x80000000u);
            unsigned long long pk = ((unsigned long long)key << 32) | (unsigned)bk;
            atomicMin(&packed[(long)b * DD + d], pk);
        }
    }
}

// ---------------------------------------------------------------------------
// host side
// ---------------------------------------------------------------------------
static inline long cdivl(long a, long b) { return (a + b - 1) / b; }

extern "C" void kernel_launch(void* const* d_in, const int* in_sizes, int n_in,
                              void* d_out, int out_size, void* d_ws, size_t ws_size,
                              hipStream_t stream)
{
    (void)in_sizes; (void)n_in; (void)out_size; (void)ws_size;
    const float* x      = (const float*)d_in[0];
    const float* cb     = (const float*)d_in[1];
    const float* pseudo = (const float*)d_in[2];
    const float* ew1    = (const float*)d_in[3];
    const float* eb1    = (const float*)d_in[4];
    const float* ew2    = (const float*)d_in[5];
    const float* eb2    = (const float*)d_in[6];
    const float* dw1    = (const float*)d_in[7];
    const float* db1    = (const float*)d_in[8];
    const float* dw2    = (const float*)d_in[9];
    const float* db2    = (const float*)d_in[10];
    float* out = (float*)d_out;

    // workspace bump allocator (256B aligned)
    size_t off = 0;
    auto alloc = [&](size_t bytes) -> char* {
        size_t o = off;
        off += (bytes + 255) & ~(size_t)255;
        return (char*)d_ws + o;
    };
    _Float16* xa16    = (_Float16*)alloc((size_t)MROWS * WDIM * 2);
    _Float16* w1t     = (_Float16*)alloc((size_t)HH * WDIM * 2);
    _Float16* w2t     = (_Float16*)alloc((size_t)(2 * ZZ) * HH * 2);
    _Float16* dw1t    = (_Float16*)alloc((size_t)HH * ZZ * 2);
    _Float16* dw2t    = (_Float16*)alloc((size_t)WDIM * HH * 2);
    _Float16* h16     = (_Float16*)alloc((size_t)MROWS * HH * 2);
    float*    encout  = (float*)   alloc((size_t)MROWS * 2 * ZZ * 4);
    _Float16* z16     = (_Float16*)alloc((size_t)BB * ZZ * 2);
    _Float16* h2_16   = (_Float16*)alloc((size_t)BB * HH * 2);
    _Float16* recon16 = (_Float16*)alloc((size_t)BB * WDIM * 2);
    _Float16* cb16    = (_Float16*)alloc((size_t)DD * KK * EE * 2);
    float*    cc      = (float*)   alloc((size_t)DD * KK * 4);
    float*    rr      = (float*)   alloc((size_t)BB * DD * 4);
    unsigned long long* packed = (unsigned long long*)alloc((size_t)BB * DD * 8);

    // ---- prep ----
    {
        long n = (long)MROWS * WDIM;
        pack_xa_f16<<<cdivl(n, 256), 256, 0, stream>>>(x, pseudo, xa16);
    }
    transpose_f32_to_f16<<<cdivl((long)WDIM * HH, 256), 256, 0, stream>>>(ew1, w1t, WDIM, HH);
    transpose_f32_to_f16<<<cdivl((long)HH * 2 * ZZ, 256), 256, 0, stream>>>(ew2, w2t, HH, 2 * ZZ);
    transpose_f32_to_f16<<<cdivl((long)ZZ * HH, 256), 256, 0, stream>>>(dw1, dw1t, ZZ, HH);
    transpose_f32_to_f16<<<cdivl((long)HH * WDIM, 256), 256, 0, stream>>>(dw2, dw2t, HH, WDIM);
    prep_codebook<<<cdivl((long)DD * KK, 256), 256, 0, stream>>>(cb, cb16, cc);
    init_packed<<<cdivl((long)BB * DD, 256), 256, 0, stream>>>(packed);

    // ---- encoder ----
    // h = relu(xa @ W1 + b1): M=912, N=512, K=4096
    gemm_wmma_f16<<<dim3((HH / 64 + 3) / 4, MROWS / 16), 128, 0, stream>>>(
        xa16, WDIM, w1t, WDIM, eb1, nullptr, h16, HH, MROWS, HH, WDIM, 1);
    // enc_out = h @ W2 + b2: M=912, N=128, K=512
    gemm_wmma_f16<<<dim3((2 * ZZ / 64 + 3) / 4, MROWS / 16), 128, 0, stream>>>(
        h16, HH, w2t, HH, eb2, encout, nullptr, 2 * ZZ, MROWS, 2 * ZZ, HH, 0);

    // ---- reparameterize + mu/logvar outputs ----
    reparam<<<cdivl((long)MROWS * ZZ, 256), 256, 0, stream>>>(encout, out, z16);

    // ---- decoder ----
    // h2 = relu(z @ dW1 + db1): M=512, N=512, K=64
    gemm_wmma_f16<<<dim3((HH / 64 + 3) / 4, BB / 16), 128, 0, stream>>>(
        z16, ZZ, dw1t, ZZ, db1, nullptr, h2_16, HH, BB, HH, ZZ, 1);
    // recon = h2 @ dW2 + db2: M=512, N=4096, K=512
    gemm_wmma_f16<<<dim3((WDIM / 64 + 3) / 4, BB / 16), 128, 0, stream>>>(
        h2_16, HH, dw2t, HH, db2, out + OFF_RECON, recon16, WDIM, BB, WDIM, HH, 0);

    // ---- codebook distances + fused argmin ----
    prep_row_norms<<<cdivl((long)BB * DD, 256), 256, 0, stream>>>(out + OFF_RECON, rr);

    // 65536 tiles (16b x 64k x 256d), 4 waves per block of 128 threads
    long dtiles = (long)DD * (BB / 16) * (KK / 64);
    dist_wmma<<<(unsigned)(dtiles / 4), 128, 0, stream>>>(
        recon16, cb16, rr, cc, out + OFF_D, packed);

    finalize_idx<<<cdivl((long)BB * DD, 256), 256, 0, stream>>>(packed, out + OFF_IDX);
}
